// QuantizedLinear_67843303408242
// MI455X (gfx1250) — matrix operand order repaired and checked by reading, not compiled, not run on hardware
//
#include <hip/hip_runtime.h>

// ---- types ----------------------------------------------------------------
typedef _Float16 v16h __attribute__((ext_vector_type(16)));
typedef _Float16 v8h  __attribute__((ext_vector_type(8)));
typedef _Float16 v4h  __attribute__((ext_vector_type(4)));
typedef float    v8f  __attribute__((ext_vector_type(8)));
typedef float    v4f  __attribute__((ext_vector_type(4)));
typedef int      v4i  __attribute__((ext_vector_type(4)));

// ---- problem constants ----------------------------------------------------
#define IN_F        4096
#define OUT_F       11008
#define TOKENS      16
#define LDS_STRIDE  (IN_F + 8)              // halves; +8 halves (16B) pad -> no LDS bank conflicts
#define WAVES_PB    8
#define TILES       (OUT_F / 16)            // 688
#define NBLOCKS     (TILES / WAVES_PB)      // 86 exactly, no remainder

// out[t, o] = scale * sum_k x[t,k] * (q[o,k] - zp)  + bias[o]
// A = x tile (16 x 32 f16, from LDS), B = (q - zp) tile (32 x 16 f16, from HBM, NT),
// C/D = f32 accumulator via v_wmma_f32_16x16x32_f16.
__global__ __launch_bounds__(256, 2)
void qlinear_wmma_kernel(const float* __restrict__ x,
                         const int*   __restrict__ qw,
                         const float* __restrict__ scale,
                         const float* __restrict__ zp,
                         const float* __restrict__ bias,
                         float*       __restrict__ out) {
    extern __shared__ _Float16 ldsx[];      // [TOKENS][LDS_STRIDE] f16

    const int tid = threadIdx.x;

    // ---- Phase 1: stage x (f32 -> f16) into LDS, whole block cooperatively.
    // 16*4096 floats = 65536 f32 = 16384 float4; 256 threads -> 64 float4 each.
    {
        const v4f* xv = (const v4f*)x;
        for (int idx = tid; idx < TOKENS * (IN_F / 4); idx += 256) {
            const int row = idx >> 10;          // IN_F/4 == 1024 float4 per row
            const int c4  = idx & 1023;
            v4f f = xv[idx];                    // contiguous: row*1024 + c4 == idx
            v4h h;
            h.x = (_Float16)f.x; h.y = (_Float16)f.y;
            h.z = (_Float16)f.z; h.w = (_Float16)f.w;
            *(v4h*)(ldsx + row * LDS_STRIDE + (c4 << 2)) = h;
        }
    }
    __syncthreads();

    const float scl = scale[0];                 // uniform -> s_load
    const float zpf = zp[0];

    // ---- Phase 2: one wave <-> one 16x16 output tile.
    const int lane = tid & 31;
    const int wave = tid >> 5;
    const int tile = blockIdx.x * WAVES_PB + wave;   // < 688 always
    const int oB   = tile << 4;
    const int m    = lane & 15;                 // A row (token) / B,D column
    const int hs   = lane >> 4;                 // half-wave select

    // A lane layout (16-bit A 16x32): lane holds row m, K in [16*hs .. 16*hs+15]
    const _Float16* aPtr = ldsx + m * LDS_STRIDE + hs * 16;
    // B lane layout (16-bit B 32x16): lane holds col m, K in [16*hs .. 16*hs+15]
    const int* wRow = qw + (size_t)(oB + m) * IN_F + hs * 16;

    v8f acc = {};
    #pragma unroll 4
    for (int k0 = 0; k0 < IN_F; k0 += 32) {
        // ---- A fragment from LDS: two 16B ds_load_b128
        v8h alo = *(const v8h*)(aPtr + k0);
        v8h ahi = *(const v8h*)(aPtr + k0 + 8);
        v16h a = __builtin_shufflevector(alo, ahi,
                 0,1,2,3,4,5,6,7,8,9,10,11,12,13,14,15);

        // ---- B fragment from HBM: 4x non-temporal global_load_b128 (streamed once)
        const v4i* wp = (const v4i*)(wRow + k0);
        v4i qv[4];
        qv[0] = __builtin_nontemporal_load(wp + 0);
        qv[1] = __builtin_nontemporal_load(wp + 1);
        qv[2] = __builtin_nontemporal_load(wp + 2);
        qv[3] = __builtin_nontemporal_load(wp + 3);

        // prefetch ~4 iterations (512B) ahead of this lane's weight stream
        __builtin_prefetch(wRow + k0 + 128, 0, 0);

        // dequant zero-point fold: b = f16(q - zp)
        v16h b;
        #pragma unroll
        for (int j = 0; j < 4; ++j)
            #pragma unroll
            for (int i = 0; i < 4; ++i)
                b[j * 4 + i] = (_Float16)((float)qv[j][i] - zpf);

        acc = __builtin_amdgcn_wmma_f32_16x16x32_f16(
                  /*neg_a=*/false, a, /*neg_b=*/false, b,
                  /*c_mod=*/(short)0, acc, /*reuse_a=*/false, /*reuse_b=*/false);
    }

    // ---- Epilogue: D 16x16 f32 layout: VGPR v -> token t = v + 8*hs, col = m
    const float bo = bias[oB + m];
    float* obase = out + (size_t)(hs * 8) * OUT_F + oB + m;
    #pragma unroll
    for (int v = 0; v < 8; ++v)
        obase[(size_t)v * OUT_F] = scl * acc[v] + bo;
}

extern "C" void kernel_launch(void* const* d_in, const int* in_sizes, int n_in,
                              void* d_out, int out_size, void* d_ws, size_t ws_size,
                              hipStream_t stream) {
    const float* x     = (const float*)d_in[0];
    const int*   qw    = (const int*)  d_in[1];
    const float* scale = (const float*)d_in[2];
    const float* zp    = (const float*)d_in[3];
    const float* bias  = (const float*)d_in[4];
    float*       out   = (float*)d_out;

    const size_t lds_bytes = (size_t)TOKENS * LDS_STRIDE * sizeof(_Float16); // 131,328 B
    qlinear_wmma_kernel<<<NBLOCKS, 256, lds_bytes, stream>>>(x, qw, scale, zp, bias, out);
}
